// RADDC_76570676953237
// MI455X (gfx1250) — compile-verified, matrix operand order checked by reading, not run on hardware
//
#include <hip/hip_runtime.h>
#include <hip/hip_bf16.h>

// ---------------------------------------------------------------------------
// RADDC: radially-scaled modulated deformable conv 3x3
//   B=4, C=256, H=W=64, O=256, dg=1
//   Main compute = GEMM: out[M=16384, N=256] = A[M, K=2304] x W[K, N]
//   A[m, k*256+c] = sum_4corners w_corner(pos,tap) * x_bf16[b, pix_corner, c]
// GEMM kernel: BM=32 pixels, BN=256, 8 waves; double-buffered LDS A tile,
// async-to-LDS metadata preload, bf16 WMMA with f32 accumulation.
// ---------------------------------------------------------------------------

typedef __attribute__((ext_vector_type(16))) __bf16 v16bf;
typedef __attribute__((ext_vector_type(8)))  float  v8f;

struct alignas(16) Meta {
    int   idx[4];   // clamped flat pixel indices (y*64+x) of 4 bilinear corners
    float w[4];     // bilinear weight * validity * mask[tap]
};

__device__ __forceinline__ unsigned short f2bf(float f) {
    unsigned int u = __float_as_uint(f);
    unsigned int r = u + 0x7FFFu + ((u >> 16) & 1u);   // round-nearest-even
    return (unsigned short)(r >> 16);
}
__device__ __forceinline__ float bf2f(unsigned int lo16) {
    return __uint_as_float(lo16 << 16);
}

// --- Kernel 0: weight (O,C,3,3) f32 -> w2[o][tap*256+c] bf16 ----------------
__global__ __launch_bounds__(256) void raddc_w2_kernel(
        const float* __restrict__ w, unsigned short* __restrict__ w2) {
    unsigned int id = blockIdx.x * 256u + threadIdx.x;   // 256*2304 = 589824
    int c = id & 255;
    unsigned int rest = id >> 8;       // o*9 + k
    int k = rest % 9;
    int o = rest / 9;
    w2[id] = f2bf(w[((size_t)o * 256 + c) * 9 + k]);     // id = o*2304 + k*256 + c
}

// --- Kernel 1: x NCHW f32 -> xT[b][pix][c] bf16 -----------------------------
__global__ __launch_bounds__(256) void raddc_xT_kernel(
        const float* __restrict__ x, unsigned short* __restrict__ xT) {
    size_t id = (size_t)blockIdx.x * 256 + threadIdx.x;  // 4*4096*256 = 4194304
    int c = (int)(id & 255);
    size_t rest = id >> 8;                                // b*4096 + pix
    int pix = (int)(rest & 4095);
    int b   = (int)(rest >> 12);
    float v = x[(((size_t)b * 256 + c) << 12) + pix];
    xT[id] = f2bf(v);
}

// --- Kernel 2: scale conv (relu), mask conv (sigmoid), sampling metadata ----
__global__ __launch_bounds__(256) void raddc_meta_kernel(
        const float* __restrict__ x,        // [4][256][64][64]
        const float* __restrict__ w_scale,  // [1][256][3][3]
        const float* __restrict__ b_scale,  // [1]
        const float* __restrict__ w_mask,   // [9][256][3][3]
        const float* __restrict__ b_mask,   // [9]
        Meta* __restrict__ meta)            // [16384][9]
{
    int pos = blockIdx.x * 256 + threadIdx.x;   // 64 blocks -> 16384
    if (pos >= 16384) return;
    int b   = pos >> 12;
    int pix = pos & 4095;
    int y   = pix >> 6;
    int xx  = pix & 63;

    float accS = b_scale[0];
    float accM[9];
    #pragma unroll
    for (int i = 0; i < 9; ++i) accM[i] = b_mask[i];

    const float* xb = x + (size_t)b * 256 * 4096;
    for (int c = 0; c < 256; ++c) {
        const float* xp = xb + (size_t)c * 4096;
        float xv[9];
        #pragma unroll
        for (int j = 0; j < 9; ++j) {
            int dy = j / 3 - 1, dx = j % 3 - 1;
            int yy = y + dy, xw = xx + dx;
            bool ok = (yy >= 0) && (yy < 64) && (xw >= 0) && (xw < 64);
            xv[j] = ok ? xp[yy * 64 + xw] : 0.0f;
        }
        const float* wsc = w_scale + c * 9;
        #pragma unroll
        for (int j = 0; j < 9; ++j) accS += wsc[j] * xv[j];
        #pragma unroll
        for (int i = 0; i < 9; ++i) {
            const float* wm = w_mask + ((size_t)i * 256 + c) * 9;
            #pragma unroll
            for (int j = 0; j < 9; ++j) accM[i] += wm[j] * xv[j];
        }
    }
    float scale = fmaxf(accS, 0.0f);   // relu; BASE_DILATION = 1

    const float cy[9] = {-1,-1,-1, 0,0,0, 1,1,1};
    const float cx[9] = {-1, 0, 1,-1,0,1,-1,0,1};
    #pragma unroll
    for (int k = 0; k < 9; ++k) {
        float mk = 1.0f / (1.0f + __expf(-accM[k]));         // sigmoid
        float ys = (float)y  - 1.0f + cy[k] * (1.0f + scale);
        float xs = (float)xx - 1.0f + cx[k] * (1.0f + scale);
        float y0f = floorf(ys), x0f = floorf(xs);
        float wy = ys - y0f, wx = xs - x0f;
        int y0 = (int)y0f, x0 = (int)x0f;
        int   yi[2]  = {y0, y0 + 1};
        int   xi[2]  = {x0, x0 + 1};
        float wyv[2] = {1.0f - wy, wy};
        float wxv[2] = {1.0f - wx, wx};
        Meta m;
        #pragma unroll
        for (int a = 0; a < 2; ++a) {
            #pragma unroll
            for (int bb = 0; bb < 2; ++bb) {
                int j = a * 2 + bb;
                bool ok = (yi[a] >= 0) && (yi[a] < 64) && (xi[bb] >= 0) && (xi[bb] < 64);
                int yc = min(max(yi[a], 0), 63);
                int xc = min(max(xi[bb], 0), 63);
                m.idx[j] = yc * 64 + xc;
                m.w[j]   = ok ? (wyv[a] * wxv[bb] * mk) : 0.0f;
            }
        }
        meta[(size_t)pos * 9 + k] = m;
    }
}

// --- A-tile build: gather 2 bf16 channels for 2 positions, combine in f32 ---
__device__ __forceinline__ void buildA(unsigned short (*__restrict__ Asb)[40],
                                       const Meta (*__restrict__ sm)[9],
                                       const unsigned short* __restrict__ xb,
                                       int t, int chunk)
{
    const int k  = chunk >> 3;
    const int ci = (chunk & 7) << 5;
    const int c0 = (t & 15) << 1;
    const int p0 = t >> 4;
    #pragma unroll
    for (int s = 0; s < 2; ++s) {
        const int p = p0 + s * 16;
        const Meta mt = sm[p][k];
        float v0 = 0.0f, v1 = 0.0f;
        #pragma unroll
        for (int j = 0; j < 4; ++j) {
            float wgt = mt.w[j];
            unsigned int u = *(const unsigned int*)
                (xb + (size_t)mt.idx[j] * 256 + ci + c0);
            v0 += wgt * bf2f(u & 0xFFFFu);
            v1 += wgt * __uint_as_float(u & 0xFFFF0000u);
        }
        unsigned int packed = (unsigned int)f2bf(v0)
                            | ((unsigned int)f2bf(v1) << 16);
        *(unsigned int*)(&Asb[p][c0]) = packed;
    }
}

// --- Kernel 3: fused gather + bf16 WMMA GEMM --------------------------------
// Block = 256 threads (8 waves). M-tile = 32 pixels, N = full 256 outputs.
// Wave w owns columns [w*32, w*32+32) as two 16x16 n-tiles x two 16-pixel
// m-subtiles -> 4 accumulators, 4 WMMAs per K-chunk.
__global__ __launch_bounds__(256) void raddc_gemm_kernel(
        const unsigned short* __restrict__ xT,   // [4][4096][256] bf16
        const unsigned short* __restrict__ w2,   // [256][2304]    bf16
        const float* __restrict__ bias,          // [256]
        const Meta* __restrict__ meta,           // [16384][9]
        float* __restrict__ out)                 // [4][256][4096]
{
    __shared__ Meta           smeta[32][9];          // 9216 B
    __shared__ unsigned short As[2][32][40];         // double-buffered, 5120 B

    const int t       = threadIdx.x;
    const int tileM   = blockIdx.x;                  // 0..511
    const int b       = tileM >> 7;                  // 128 tiles per batch
    const int pixBase = (tileM & 127) << 5;          // 32 consecutive pixels
    const unsigned short* xb = xT + ((size_t)b << 20);   // b*4096*256

    // Async preload of sampling metadata straight into LDS: 576 x b128
    {
        const char*  g     = (const char*)(meta + (size_t)(b * 4096 + pixBase) * 9);
        unsigned int sbase = (unsigned int)(unsigned long long)(&smeta[0][0]);
        for (int i = t; i < 32 * 9 * 2; i += 256) {
            unsigned long long ga = (unsigned long long)(g + i * 16);
            unsigned int       la = sbase + i * 16;
            asm volatile("global_load_async_to_lds_b128 %0, %1, off"
                         :: "v"(la), "v"(ga) : "memory");
        }
        asm volatile("s_wait_asynccnt 0x0" ::: "memory");
    }
    __syncthreads();

    const int lane = t & 31;
    const int wave = t >> 5;
    const int n    = lane & 15;     // output column within 16x16 tile / A row
    const int hi   = lane >> 4;     // K-half selector per 16-bit WMMA layout

    const int o0 = wave * 32 + n;
    const int o1 = o0 + 16;
    v8f acc00 = {}, acc01 = {}, acc10 = {}, acc11 = {};

    buildA(As[0], smeta, xb, t, 0);
    __syncthreads();

    for (int chunk = 0; chunk < 72; ++chunk) {       // 9 taps x 8 c-chunks
        const int cur = chunk & 1;
        if (chunk + 1 < 72) buildA(As[cur ^ 1], smeta, xb, t, chunk + 1);

        // Fragments (16-bit WMMA layout: lanes 0-15 K0-7/K16-23,
        // lanes 16-31 K8-15/K24-31)
        union { uint4 q[2]; v16bf v; } a0, a1, bfr0, bfr1;
        const uint4* ap0 = (const uint4*)(&As[cur][n][hi * 8]);
        a0.q[0] = ap0[0];
        a0.q[1] = ap0[2];
        const uint4* ap1 = (const uint4*)(&As[cur][16 + n][hi * 8]);
        a1.q[0] = ap1[0];
        a1.q[1] = ap1[2];

        const int kb = (chunk >> 3) * 256 + ((chunk & 7) << 5);
        const uint4* bp0 = (const uint4*)(w2 + (size_t)o0 * 2304 + kb + hi * 8);
        bfr0.q[0] = bp0[0];
        bfr0.q[1] = bp0[2];
        const uint4* bp1 = (const uint4*)(w2 + (size_t)o1 * 2304 + kb + hi * 8);
        bfr1.q[0] = bp1[0];
        bfr1.q[1] = bp1[2];
        __builtin_prefetch(bp0 + 4, 0, 0);   // next K-chunk of this weight row

        acc00 = __builtin_amdgcn_wmma_f32_16x16x32_bf16(
                    false, a0.v, false, bfr0.v, (short)0, acc00, false, false);
        acc01 = __builtin_amdgcn_wmma_f32_16x16x32_bf16(
                    false, a0.v, false, bfr1.v, (short)0, acc01, false, false);
        acc10 = __builtin_amdgcn_wmma_f32_16x16x32_bf16(
                    false, a1.v, false, bfr0.v, (short)0, acc10, false, false);
        acc11 = __builtin_amdgcn_wmma_f32_16x16x32_bf16(
                    false, a1.v, false, bfr1.v, (short)0, acc11, false, false);

        __syncthreads();
    }

    // ---- epilogue: C/D layout -> out[b][o][pix] + bias ----
    const float bias0 = bias[o0];
    const float bias1 = bias[o1];
    float* ob0 = out + (size_t)(b * 256 + o0) * 4096 + pixBase;
    float* ob1 = out + (size_t)(b * 256 + o1) * 4096 + pixBase;
    #pragma unroll
    for (int r = 0; r < 8; ++r) {
        int m = r + hi * 8;      // VGPR r: M=r (lanes 0-15) / M=r+8 (lanes 16-31)
        ob0[m]      = acc00[r] + bias0;
        ob1[m]      = acc01[r] + bias1;
        ob0[m + 16] = acc10[r] + bias0;
        ob1[m + 16] = acc11[r] + bias1;
    }
}

// ---------------------------------------------------------------------------
extern "C" void kernel_launch(void* const* d_in, const int* in_sizes, int n_in,
                              void* d_out, int out_size, void* d_ws, size_t ws_size,
                              hipStream_t stream) {
    const float* x       = (const float*)d_in[0];   // 4*256*64*64
    const float* weight  = (const float*)d_in[1];   // 256*256*3*3
    const float* bias    = (const float*)d_in[2];   // 256
    const float* w_scale = (const float*)d_in[3];   // 1*256*3*3
    const float* b_scale = (const float*)d_in[4];   // 1
    const float* w_mask  = (const float*)d_in[5];   // 9*256*3*3
    const float* b_mask  = (const float*)d_in[6];   // 9
    float* out = (float*)d_out;                     // 4*256*64*64

    char* ws = (char*)d_ws;
    unsigned short* xT = (unsigned short*)ws;                       // 8,388,608 B
    unsigned short* w2 = (unsigned short*)(ws + (8u << 20));        // 1,179,648 B
    Meta*           mt = (Meta*)(ws + (8u << 20) + 1179648);        // 4,718,592 B
    (void)ws_size; (void)in_sizes; (void)n_in; (void)out_size;

    raddc_w2_kernel  <<<2304,  256, 0, stream>>>(weight, w2);
    raddc_xT_kernel  <<<16384, 256, 0, stream>>>(x, xT);
    raddc_meta_kernel<<<64,    256, 0, stream>>>(x, w_scale, b_scale, w_mask, b_mask, mt);
    raddc_gemm_kernel<<<512,   256, 0, stream>>>(xT, w2, bias, mt, out);
}
